// SelfAttention_33852932227438
// MI455X (gfx1250) — matrix operand order repaired and checked by reading, not compiled, not run on hardware
//
#include <hip/hip_runtime.h>
#include <hip/hip_bf16.h>

// ---------------------------------------------------------------------------
// Self-attention, bf16 WMMA path for gfx1250 (MI455X).
// S=2048, B=4, D=1024, H=16, HD=64.  M = S*B = 8192.
// d_out = [ out (8192*1024 f32) | weights (4*2048*2048 f32) ]
// ---------------------------------------------------------------------------

typedef __attribute__((ext_vector_type(16))) __bf16 v16bf;
typedef __attribute__((ext_vector_type(8)))  float  v8f;

union BF16x16 {
    uint4          q[2];
    v16bf          v;
    unsigned short u[16];
};

__device__ __forceinline__ unsigned short f2bf(float f) {
    unsigned u = __builtin_bit_cast(unsigned, f);
    u += 0x7FFFu + ((u >> 16) & 1u);   // round-to-nearest-even
    return (unsigned short)(u >> 16);
}

__device__ __forceinline__ v8f wmma_bf16(v16bf a, v16bf b, v8f c) {
    return __builtin_amdgcn_wmma_f32_16x16x32_bf16(
        /*neg_a=*/false, a, /*neg_b=*/false, b,
        /*c_mod=*/(short)0, c, /*reuse_a=*/false, /*reuse_b=*/false);
}

// reductions across the 16 lanes of each half-wave (C-layout columns)
__device__ __forceinline__ float half16_max(float v) {
    v = fmaxf(v, __shfl_xor(v, 1, 32));
    v = fmaxf(v, __shfl_xor(v, 2, 32));
    v = fmaxf(v, __shfl_xor(v, 4, 32));
    v = fmaxf(v, __shfl_xor(v, 8, 32));
    return v;
}
__device__ __forceinline__ float half16_sum(float v) {
    v += __shfl_xor(v, 1, 32);
    v += __shfl_xor(v, 2, 32);
    v += __shfl_xor(v, 4, 32);
    v += __shfl_xor(v, 8, 32);
    return v;
}

// ---- async global -> LDS staging (gfx1250 ASYNCcnt path), with fallback ----
#if __has_builtin(__builtin_amdgcn_global_load_async_to_lds_b128)
#define HAS_ASYNC_LDS 1
#else
#define HAS_ASYNC_LDS 0
#endif

__device__ __forceinline__ void load16_to_lds(unsigned short* lds_dst,
                                              const unsigned short* gsrc) {
#if HAS_ASYNC_LDS
    typedef int v4i __attribute__((vector_size(16)));
    typedef __attribute__((address_space(1))) v4i* gptr_t;   // global (AS1)
    typedef __attribute__((address_space(3))) v4i* lptr_t;   // LDS    (AS3)
    __builtin_amdgcn_global_load_async_to_lds_b128(
        (gptr_t)(void*)const_cast<unsigned short*>(gsrc),
        (lptr_t)(void*)lds_dst, 0, 0);
#else
    *(uint4*)lds_dst = *(const uint4*)gsrc;
#endif
}

__device__ __forceinline__ void async_wait0() {
#if HAS_ASYNC_LDS
#if __has_builtin(__builtin_amdgcn_s_wait_asynccnt)
    __builtin_amdgcn_s_wait_asynccnt(0);
#else
    asm volatile("s_wait_asynccnt 0x0" ::: "memory");
#endif
#endif
}

// Workspace layout (bf16/ushort elements):
#define OFF_XB 0u           // x            8192x1024
#define OFF_WQ 8388608u     // w_q          1024x1024
#define OFF_WK 9437184u
#define OFF_WV 10485760u
#define OFF_WO 11534336u
#define OFF_Q  12582912u    // Q  [B,H,S,HD] (pre-scaled by 1/sqrt(HD))
#define OFF_K  20971520u    // K  [B,H,S,HD]
#define OFF_V  29360128u    // V^T [B,H,HD,S]  (transposed for K-contiguous B tiles)
#define OFF_AO 37748736u    // attn out, [M=8192, D=1024], row = s*B+b

// ---------------------------------------------------------------------------
// Kernel 0: f32 -> bf16 conversion of x and the four weight matrices.
// ---------------------------------------------------------------------------
__global__ void cvt_kernel(const float* __restrict__ x,  const float* __restrict__ wq,
                           const float* __restrict__ wk, const float* __restrict__ wv,
                           const float* __restrict__ wo, unsigned short* __restrict__ ws) {
    const unsigned total = 8388608u + 4u * 1048576u;
    unsigned stride = gridDim.x * blockDim.x;
    for (unsigned i = blockIdx.x * blockDim.x + threadIdx.x; i < total; i += stride) {
        float f;
        if (i < 8388608u) {
            f = x[i];
        } else {
            unsigned j = i - 8388608u;
            unsigned w = j >> 20, idx = j & 1048575u;
            const float* p = (w == 0) ? wq : (w == 1) ? wk : (w == 2) ? wv : wo;
            f = p[idx];
        }
        ws[i] = f2bf(f);
    }
}

// ---------------------------------------------------------------------------
// Kernel 1: QKV projection.  out[m,n] = dot(x[m,:], w[n,:]) + bias[n]
// grid = (M/128, D/64, 3);  block = 128 (4 waves); wave -> 16x64 tile.
// z==0 -> Q [B,H,S,HD] scaled by 0.125;  z==1 -> K [B,H,S,HD];
// z==2 -> V stored TRANSPOSED [B,H,HD,S].
// ---------------------------------------------------------------------------
__global__ void qkv_kernel(unsigned short* __restrict__ ws,
                           const float* __restrict__ bq,
                           const float* __restrict__ bk,
                           const float* __restrict__ bv) {
    const unsigned short* xb = ws + OFF_XB;
    const int z = blockIdx.z;
    const unsigned short* wb = ws + OFF_WQ + (size_t)z * 1048576u;
    const float* bias = (z == 0) ? bq : (z == 1) ? bk : bv;
    unsigned short* outb = ws + OFF_Q + (size_t)z * 8388608u;

    const int lane = threadIdx.x & 31, wvid = threadIdx.x >> 5;
    const int l16 = lane & 15, lh = lane >> 4;
    const int m0 = blockIdx.x * 128 + wvid * 16;
    const int n0 = blockIdx.y * 64;

    v8f acc[4] = {};
    const unsigned short* arow = xb + (size_t)(m0 + l16) * 1024u;

    for (int kk = 0; kk < 1024; kk += 32) {
        BF16x16 a, bm[4];
        a.q[0] = *(const uint4*)(arow + kk + lh * 16);
        a.q[1] = *(const uint4*)(arow + kk + lh * 16 + 8);
        __builtin_prefetch(arow + kk + 512, 0, 1);
#pragma unroll
        for (int j = 0; j < 4; ++j) {
            const unsigned short* brow =
                wb + (size_t)(n0 + j * 16 + l16) * 1024u + kk + lh * 16;
            bm[j].q[0] = *(const uint4*)(brow);
            bm[j].q[1] = *(const uint4*)(brow + 8);
        }
#pragma unroll
        for (int j = 0; j < 4; ++j)
            acc[j] = wmma_bf16(a.v, bm[j].v, acc[j]);
    }

    const float scale = (z == 0) ? 0.125f : 1.0f;
#pragma unroll
    for (int j = 0; j < 4; ++j) {
        int n = n0 + j * 16 + l16;
        float bval = bias[n];
        int h = n >> 6, hd = n & 63;
#pragma unroll
        for (int r = 0; r < 8; ++r) {
            int m = m0 + r + 8 * lh;          // C layout: row = r + 8*half
            int s = m >> 2, bb = m & 3;       // m = s*B + b, B=4
            float val = (acc[j][r] + bval) * scale;
            if (z == 2)  // V^T: [B,H,HD,S]
                outb[((size_t)((bb * 16 + h) * 64 + hd)) * 2048u + s] = f2bf(val);
            else         // Q/K: [B,H,S,HD]
                outb[((size_t)((bb * 16 + h) * 2048 + s)) * 64u + hd] = f2bf(val);
        }
    }
}

// ---------------------------------------------------------------------------
// Kernel 2: attention.  grid = (B, S/64); block = 128 (4 waves);
// each wave owns 16 query rows; all 4 waves share (b, h) so K / V^T chunks
// are staged once per block in LDS via async global->LDS loads.
// Pass A: online softmax stats (max, sumexp) via QK^T WMMA tiles.
// Pass B: normalized probs -> weights output (+= p/H) and P@V via LDS relayout.
// ---------------------------------------------------------------------------
__global__ void attn_kernel(unsigned short* __restrict__ ws,
                            float* __restrict__ out_w /* = d_out + 8388608 */) {
    __shared__ unsigned short kct[32 * 64];      // K chunk  [32 keys][64 hd]  4KB
    __shared__ unsigned short vct[64 * 32];      // V^T chunk [64 hd][32 keys] 4KB
    __shared__ unsigned short plds[4][16 * 32];  // per-wave P staging         4KB

    const int tid  = threadIdx.x;
    const int lane = tid & 31, wvid = tid >> 5;
    const int l16 = lane & 15, lh = lane >> 4;
    const int b  = blockIdx.x;
    const int q0 = blockIdx.y * 64 + wvid * 16;

    const unsigned short* qb = ws + OFF_Q;
    const unsigned short* kb = ws + OFF_K;
    const unsigned short* vb = ws + OFF_V;
    unsigned short*       ab = ws + OFF_AO;
    float* wout = out_w + (size_t)b * 2048u * 2048u;

    for (int h = 0; h < 16; ++h) {
        const size_t bh = (size_t)(b * 16 + h) * 2048u * 64u;
        const unsigned short* qbh = qb + bh;
        const unsigned short* kbh = kb + bh;
        const unsigned short* vbh = vb + bh;   // V^T: [64 hd][2048 keys]

        // Q tile (16 x 64) in A layout, two 16x32 chunks, reused all pass.
        BF16x16 qa[2];
        const unsigned short* qrow = qbh + (size_t)(q0 + l16) * 64u;
#pragma unroll
        for (int c = 0; c < 2; ++c) {
            qa[c].q[0] = *(const uint4*)(qrow + c * 32 + lh * 16);
            qa[c].q[1] = *(const uint4*)(qrow + c * 32 + lh * 16 + 8);
        }

        float mrow[8], lrow[8];
#pragma unroll
        for (int r = 0; r < 8; ++r) { mrow[r] = -1e30f; lrow[r] = 0.0f; }

        // ---- Pass A: softmax statistics (32 keys per chunk) -----------
        for (int t0 = 0; t0 < 2048; t0 += 32) {
            __syncthreads();                        // WAR on kct
            // cooperative async load: K chunk = contiguous 4KB = 256 x 16B
#pragma unroll
            for (int u = 0; u < 2; ++u) {
                int unit = tid + u * 128;
                load16_to_lds(&kct[unit * 8], kbh + (size_t)t0 * 64u + unit * 8);
            }
            async_wait0();
            __syncthreads();

#pragma unroll
            for (int sub = 0; sub < 2; ++sub) {
                v8f sc = {};
#pragma unroll
                for (int c = 0; c < 2; ++c) {
                    BF16x16 kt;   // B tile: col = key, elems over hd chunk
                    const unsigned short* kr =
                        &kct[(sub * 16 + l16) * 64 + c * 32 + lh * 16];
                    kt.q[0] = *(const uint4*)(kr);
                    kt.q[1] = *(const uint4*)(kr + 8);
                    sc = wmma_bf16(qa[c].v, kt.v, sc);
                }
#pragma unroll
                for (int r = 0; r < 8; ++r) {
                    float sv   = sc[r];
                    float tmax = half16_max(sv);
                    float mn   = fmaxf(mrow[r], tmax);
                    float tsum = half16_sum(__expf(sv - mn));
                    lrow[r] = lrow[r] * __expf(mrow[r] - mn) + tsum;
                    mrow[r] = mn;
                }
            }
        }

        float il[8];
#pragma unroll
        for (int r = 0; r < 8; ++r) il[r] = 1.0f / lrow[r];

        // ---- Pass B: probs -> weights output, and P @ V ---------------
        v8f oacc[4] = {};
        for (int t0 = 0; t0 < 2048; t0 += 32) {
            __syncthreads();                        // WAR on kct/vct
#pragma unroll
            for (int u = 0; u < 2; ++u) {
                int unit = tid + u * 128;
                // K chunk: contiguous
                load16_to_lds(&kct[unit * 8], kbh + (size_t)t0 * 64u + unit * 8);
                // V^T chunk: 64 rows x 32 keys; unit -> (row, part)
                int row = unit >> 2, part = unit & 3;
                load16_to_lds(&vct[row * 32 + part * 8],
                              vbh + (size_t)row * 2048u + t0 + part * 8);
            }
            async_wait0();
            __syncthreads();

#pragma unroll
            for (int sub = 0; sub < 2; ++sub) {
                int tbase = t0 + sub * 16;
                v8f sc = {};
#pragma unroll
                for (int c = 0; c < 2; ++c) {
                    BF16x16 kt;
                    const unsigned short* kr =
                        &kct[(sub * 16 + l16) * 64 + c * 32 + lh * 16];
                    kt.q[0] = *(const uint4*)(kr);
                    kt.q[1] = *(const uint4*)(kr + 8);
                    sc = wmma_bf16(qa[c].v, kt.v, sc);
                }
#pragma unroll
                for (int r = 0; r < 8; ++r) {
                    float p   = __expf(sc[r] - mrow[r]) * il[r]; // normalized
                    int   row = r + 8 * lh;
                    plds[wvid][row * 32 + sub * 16 + l16] = f2bf(p);
                    size_t widx = (size_t)(q0 + row) * 2048u + tbase + l16;
                    float  wv   = p * 0.0625f;                   // /H
                    if (h == 0) wout[widx] = wv;
                    else        wout[widx] += wv;                 // same wave only
                }
            }
            // reload P (16x32) in A layout from LDS (in-order per wave)
            BF16x16 pa;
            const unsigned short* pr = &plds[wvid][l16 * 32];
            pa.q[0] = *(const uint4*)(pr + lh * 16);
            pa.q[1] = *(const uint4*)(pr + lh * 16 + 8);
#pragma unroll
            for (int j = 0; j < 4; ++j) {
                BF16x16 vt;   // B tile: col = hd, elems over keys (K-contig in V^T)
                const unsigned short* vr = &vct[(j * 16 + l16) * 32 + lh * 16];
                vt.q[0] = *(const uint4*)(vr);
                vt.q[1] = *(const uint4*)(vr + 8);
                oacc[j] = wmma_bf16(pa.v, vt.v, oacc[j]);
            }
        }

        // ---- store head output tile (rows q0..q0+15, cols h*64..h*64+63)
#pragma unroll
        for (int j = 0; j < 4; ++j)
#pragma unroll
            for (int r = 0; r < 8; ++r) {
                int row = r + 8 * lh;
                int s   = q0 + row;
                ab[(size_t)(s * 4 + b) * 1024u + h * 64 + j * 16 + l16] =
                    f2bf(oacc[j][r]);
            }
    }
}

// ---------------------------------------------------------------------------
// Kernel 3: output projection.  out[m,n] = dot(ao[m,:], w_o[n,:]) + b_o[n]
// ---------------------------------------------------------------------------
__global__ void oproj_kernel(const unsigned short* __restrict__ ws,
                             const float* __restrict__ bo,
                             float* __restrict__ out) {
    const unsigned short* abuf = ws + OFF_AO;
    const unsigned short* wb   = ws + OFF_WO;

    const int lane = threadIdx.x & 31, wvid = threadIdx.x >> 5;
    const int l16 = lane & 15, lh = lane >> 4;
    const int m0 = blockIdx.x * 128 + wvid * 16;
    const int n0 = blockIdx.y * 64;

    v8f acc[4] = {};
    const unsigned short* arow = abuf + (size_t)(m0 + l16) * 1024u;

    for (int kk = 0; kk < 1024; kk += 32) {
        BF16x16 a, bm[4];
        a.q[0] = *(const uint4*)(arow + kk + lh * 16);
        a.q[1] = *(const uint4*)(arow + kk + lh * 16 + 8);
        __builtin_prefetch(arow + kk + 512, 0, 1);
#pragma unroll
        for (int j = 0; j < 4; ++j) {
            const unsigned short* brow =
                wb + (size_t)(n0 + j * 16 + l16) * 1024u + kk + lh * 16;
            bm[j].q[0] = *(const uint4*)(brow);
            bm[j].q[1] = *(const uint4*)(brow + 8);
        }
#pragma unroll
        for (int j = 0; j < 4; ++j)
            acc[j] = wmma_bf16(a.v, bm[j].v, acc[j]);
    }
#pragma unroll
    for (int j = 0; j < 4; ++j) {
        int n = n0 + j * 16 + l16;
        float bval = bo[n];
#pragma unroll
        for (int r = 0; r < 8; ++r) {
            int m = m0 + r + 8 * lh;
            out[(size_t)m * 1024u + n] = acc[j][r] + bval;
        }
    }
}

// ---------------------------------------------------------------------------
extern "C" void kernel_launch(void* const* d_in, const int* in_sizes, int n_in,
                              void* d_out, int out_size, void* d_ws, size_t ws_size,
                              hipStream_t stream) {
    (void)in_sizes; (void)n_in; (void)out_size; (void)ws_size;
    const float* x  = (const float*)d_in[0];
    const float* wq = (const float*)d_in[1];
    const float* bq = (const float*)d_in[2];
    const float* wk = (const float*)d_in[3];
    const float* bk = (const float*)d_in[4];
    const float* wv = (const float*)d_in[5];
    const float* bv = (const float*)d_in[6];
    const float* wo = (const float*)d_in[7];
    const float* bo = (const float*)d_in[8];

    unsigned short* ws = (unsigned short*)d_ws;
    float* out = (float*)d_out;

    cvt_kernel  <<<dim3(512),       dim3(256), 0, stream>>>(x, wq, wk, wv, wo, ws);
    qkv_kernel  <<<dim3(64, 16, 3), dim3(128), 0, stream>>>(ws, bq, bk, bv);
    attn_kernel <<<dim3(4, 32),     dim3(128), 0, stream>>>(ws, out + 8388608u);
    oproj_kernel<<<dim3(64, 16),    dim3(128), 0, stream>>>(ws, bo, out);
}
